// VoxelRasterizer_49220325212758
// MI455X (gfx1250) — compile-verified, matrix operand order checked by reading, not run on hardware
//
#include <hip/hip_runtime.h>
#include <stdint.h>

// ---------------------------------------------------------------------------
// VoxelRasterizer for MI455X (gfx1250, wave32)
//   k1: mean ray direction (block reduce)
//   k2: sort keys = morton + (positions @ mean_d)*1e-6 via v_wmma_f32_16x16x32_f16
//   k3: bitonic sort of 4096 keys (one block, 32KB LDS)
//   k4: gather into packed voxel records {bmin.xyz,bmax.x | bmax.yz,dens,pad} + sigmoid colors
//   k5: render: TDM stages 128KB voxel table into LDS, one wave per ray,
//       ballot-compacted hit list, rank sort, serial front-to-back composite
// ---------------------------------------------------------------------------

#define MAX_HITS        128
#define RAYS_PER_BLOCK  8
#define S_SAMPLES_F     8.0f
#define T_MIN_C         0.0001f

typedef __attribute__((ext_vector_type(16))) _Float16     v16h;
typedef __attribute__((ext_vector_type(8)))  float        v8f;
typedef __attribute__((ext_vector_type(4)))  unsigned int u32x4;
typedef __attribute__((ext_vector_type(8)))  int          i32x8;
typedef __attribute__((ext_vector_type(4)))  int          i32x4;

__device__ __forceinline__ unsigned int ballot32(bool p) {
#if __has_builtin(__builtin_amdgcn_ballot_w32)
  return __builtin_amdgcn_ballot_w32(p);
#else
  return (unsigned int)__ballot(p);
#endif
}

// ---------------------------------------------------------------- k1: mean dir
__global__ __launch_bounds__(256) void vr_meand(const float* __restrict__ rd,
                                                float* __restrict__ meand, int N) {
  __shared__ float sx[256], sy[256], sz[256];
  int t = threadIdx.x;
  float x = 0.f, y = 0.f, z = 0.f;
  for (int i = t; i < N; i += 256) {
    x += rd[i * 3 + 0]; y += rd[i * 3 + 1]; z += rd[i * 3 + 2];
  }
  sx[t] = x; sy[t] = y; sz[t] = z;
  __syncthreads();
  for (int s = 128; s > 0; s >>= 1) {
    if (t < s) { sx[t] += sx[t + s]; sy[t] += sy[t + s]; sz[t] += sz[t + s]; }
    __syncthreads();
  }
  if (t == 0) {
    meand[0] = sx[0] / (float)N;
    meand[1] = sy[0] / (float)N;
    meand[2] = sz[0] / (float)N;
  }
}

// ------------------------------------------------- k2: sort keys via WMMA dots
// One wave computes dots for 16 voxels: D = A(16x32 f16) x B(32x16 f16), where
// A rows = [px,py,pz,0...] and B column 0 = mean_d. 16-bit A layout: lanes 0-15
// hold K=0..7 in VGPR0-3 (2 halves/VGPR), so a[0..2] = px,py,pz on lanes 0-15.
__global__ __launch_bounds__(256) void vr_keys(const float* __restrict__ positions,
                                               const int* __restrict__ morton,
                                               const float* __restrict__ meand,
                                               float* __restrict__ keys, int V) {
  int gtid = blockIdx.x * blockDim.x + threadIdx.x;
  int wave = gtid >> 5;
  int lane = threadIdx.x & 31;
  int base = wave * 16;
  if (base >= V) return;  // uniform per wave; EXEC stays all-ones for WMMA

  v16h a = {};
  v16h b = {};
  if (lane < 16) {
    int v = base + lane;
    if (v < V) {
      a[0] = (_Float16)positions[v * 3 + 0];
      a[1] = (_Float16)positions[v * 3 + 1];
      a[2] = (_Float16)positions[v * 3 + 2];
    }
    b[0] = (_Float16)meand[0];
    b[1] = (_Float16)meand[1];
    b[2] = (_Float16)meand[2];
  }
  v8f c = {};
  c = __builtin_amdgcn_wmma_f32_16x16x32_f16(false, a, false, b,
                                             (short)0, c, false, false);
  // D layout: lane n (0-15) holds D[r][n] in VGPR r, lane 16+n holds D[8+r][n].
  // Column 0 (the dots) lives on lanes 0 and 16.
  if (lane == 0) {
    for (int r = 0; r < 8; ++r) {
      int v = base + r;
      if (v < V) keys[v] = (float)morton[v] + c[r] * 1e-6f;
    }
  } else if (lane == 16) {
    for (int r = 0; r < 8; ++r) {
      int v = base + 8 + r;
      if (v < V) keys[v] = (float)morton[v] + c[r] * 1e-6f;
    }
  }
}

// ------------------------------------------------------- k3: bitonic sort 4096
__global__ __launch_bounds__(1024) void vr_sort(const float* __restrict__ keys,
                                                int* __restrict__ order, int V) {
  __shared__ float sk[4096];
  __shared__ int   sv[4096];
  int t = threadIdx.x;
  for (int i = t; i < 4096; i += 1024) {
    sk[i] = (i < V) ? keys[i] : 3.0e38f;
    sv[i] = i;
  }
  __syncthreads();
  for (int len = 2; len <= 4096; len <<= 1) {
    for (int j = len >> 1; j > 0; j >>= 1) {
      for (int p = t; p < 2048; p += 1024) {
        int i = ((p & ~(j - 1)) << 1) | (p & (j - 1));
        int k = i | j;
        bool asc = ((i & len) == 0);
        float a = sk[i], bb = sk[k];
        bool sw = asc ? (a > bb) : (a < bb);
        if (sw) {
          sk[i] = bb; sk[k] = a;
          int tmp = sv[i]; sv[i] = sv[k]; sv[k] = tmp;
        }
      }
      __syncthreads();
    }
  }
  for (int i = t; i < V; i += 1024) order[i] = sv[i];
}

// ----------------------------------------------- k4: gather + activation preco
__global__ void vr_gather(const int* __restrict__ order,
                          const float* __restrict__ positions,
                          const float* __restrict__ sizes,
                          const float* __restrict__ densities,
                          const float* __restrict__ colors,
                          float* __restrict__ packed,
                          float* __restrict__ colsig, int V) {
  int s = blockIdx.x * blockDim.x + threadIdx.x;
  if (s >= V) return;
  int v = order[s];
  float h  = sizes[v] * 0.5f;
  float px = positions[v * 3 + 0];
  float py = positions[v * 3 + 1];
  float pz = positions[v * 3 + 2];
  packed[s * 8 + 0] = px - h;
  packed[s * 8 + 1] = py - h;
  packed[s * 8 + 2] = pz - h;
  packed[s * 8 + 3] = px + h;
  packed[s * 8 + 4] = py + h;
  packed[s * 8 + 5] = pz + h;
  packed[s * 8 + 6] = expf(densities[v]);
  packed[s * 8 + 7] = 0.0f;
  for (int c = 0; c < 3; ++c)
    colsig[s * 3 + c] = 1.0f / (1.0f + expf(-colors[v * 27 + c]));
}

// --------------------------------------------------------------- k5: renderer
__global__ __launch_bounds__(256) void vr_render(const float* __restrict__ packed,
                                                 const float* __restrict__ colsig,
                                                 const float* __restrict__ ro,
                                                 const float* __restrict__ rd,
                                                 float* __restrict__ out,
                                                 int V, int N) {
  extern __shared__ float smem[];
  float* vox = smem;                                   // V*8 floats (128KB @4096)
  float* htn = vox + (size_t)V * 8;                    // raw hit lists
  float* htf = htn + RAYS_PER_BLOCK * MAX_HITS;
  int*   hid = (int*)(htf + RAYS_PER_BLOCK * MAX_HITS);
  float* stn = (float*)(hid + RAYS_PER_BLOCK * MAX_HITS);  // sorted hit lists
  float* stf = stn + RAYS_PER_BLOCK * MAX_HITS;
  int*   sid = (int*)(stf + RAYS_PER_BLOCK * MAX_HITS);

  // ---- Stage voxel table into LDS via the Tensor Data Mover (one 1-D tile).
#if __has_builtin(__builtin_amdgcn_tensor_load_to_lds)
  if (threadIdx.x == 0) {
    unsigned long long ga = (unsigned long long)(uintptr_t)packed;
    unsigned int lds_base = (unsigned int)(uintptr_t)vox;  // addr[31:0] = LDS offset
    unsigned int nelem = (unsigned int)V * 8u;             // f32 elements
    u32x4 g0;
    g0[0] = 1u;                                            // count=1, user mode
    g0[1] = lds_base;                                      // lds_addr
    g0[2] = (unsigned int)(ga & 0xFFFFFFFFull);            // global_addr[31:0]
    g0[3] = (unsigned int)((ga >> 32) & 0x1FFFFFFull)      // global_addr[56:32]
          | (2u << 30);                                    // type=2 ("image")
    i32x8 g1;
    g1[0] = (int)(2u << 16);                               // data_size=4B
    g1[1] = (int)((nelem & 0xFFFFu) << 16);                // tensor_dim0[15:0]
    g1[2] = (int)((nelem >> 16) | (1u << 16));             // dim0[31:16], dim1=1
    g1[3] = (int)((nelem & 0xFFFFu) << 16);                // tile_dim0 = nelem
    g1[4] = 1;                                             // tile_dim1=1, tile_dim2=0
    g1[5] = (int)nelem;                                    // tensor_dim0_stride
    g1[6] = 0;
    g1[7] = 0;
    i32x4 gz = {0, 0, 0, 0};
#if defined(__clang_major__) && (__clang_major__ >= 23)
    i32x8 gz8 = {0, 0, 0, 0, 0, 0, 0, 0};
    __builtin_amdgcn_tensor_load_to_lds(g0, g1, gz, gz, gz8, 0);
#else
    __builtin_amdgcn_tensor_load_to_lds(g0, g1, gz, gz, 0);
#endif
    __builtin_amdgcn_s_wait_tensorcnt(0);
  }
#else
  for (int i = threadIdx.x; i < V * 8; i += blockDim.x) vox[i] = packed[i];
#endif
  __syncthreads();

  int wave = threadIdx.x >> 5;
  int lane = threadIdx.x & 31;
  int ray  = blockIdx.x * RAYS_PER_BLOCK + wave;
  bool active = ray < N;

  float ox = 0.f, oy = 0.f, oz = 0.f, ix = 0.f, iy = 0.f, iz = 0.f;
  if (active) {
    ox = ro[ray * 3 + 0]; oy = ro[ray * 3 + 1]; oz = ro[ray * 3 + 2];
    float dx = rd[ray * 3 + 0], dy = rd[ray * 3 + 1], dz = rd[ray * 3 + 2];
    ix = 1.0f / (dx + 1e-8f);
    iy = 1.0f / (dy + 1e-8f);
    iz = 1.0f / (dz + 1e-8f);
  }

  // ---- Intersect all voxels, wave32 ballot-compacted hit list in LDS.
  int cnt = 0;
  if (active) {
    for (int vb = 0; vb < V; vb += 32) {
      int v = vb + lane;
      const float4* vp = (const float4*)(vox + (size_t)v * 8);
      float4 p0 = vp[0];   // bmin.xyz, bmax.x
      float4 p1 = vp[1];   // bmax.y, bmax.z, dens, pad
      float t1x = (p0.x - ox) * ix, t2x = (p0.w - ox) * ix;
      float t1y = (p0.y - oy) * iy, t2y = (p1.x - oy) * iy;
      float t1z = (p0.z - oz) * iz, t2z = (p1.y - oz) * iz;
      float tn3 = fmaxf(fmaxf(fminf(t1x, t2x), fminf(t1y, t2y)), fminf(t1z, t2z));
      float tf3 = fminf(fminf(fmaxf(t1x, t2x), fmaxf(t1y, t2y)), fmaxf(t1z, t2z));
      bool hit = (tn3 <= tf3) && (tf3 > 0.0f);
      float tn = fmaxf(tn3, 0.0f);
      unsigned int m = ballot32(hit);
      if (hit) {
        int slot = cnt + __popc(m & ((1u << lane) - 1u));
        if (slot < MAX_HITS) {
          htn[wave * MAX_HITS + slot] = tn;
          htf[wave * MAX_HITS + slot] = tf3;
          hid[wave * MAX_HITS + slot] = v;
        }
      }
      cnt += __popc(m);
    }
    if (cnt > MAX_HITS) cnt = MAX_HITS;
  }
  __syncthreads();

  // ---- Rank sort (stable: tie-break by globally-sorted index order).
  if (active) {
    for (int i = lane; i < cnt; i += 32) {
      float ti = htn[wave * MAX_HITS + i];
      int r = 0;
      for (int j = 0; j < cnt; ++j) {
        float tj = htn[wave * MAX_HITS + j];
        r += (tj < ti) || ((tj == ti) && (j < i));
      }
      stn[wave * MAX_HITS + r] = ti;
      stf[wave * MAX_HITS + r] = htf[wave * MAX_HITS + i];
      sid[wave * MAX_HITS + r] = hid[wave * MAX_HITS + i];
    }
  }
  __syncthreads();

  // ---- Front-to-back composite with T_MIN early termination (lane 0).
  if (active && lane == 0) {
    float T = 1.0f, r = 0.f, g = 0.f, b = 0.f, dep = 0.f;
    for (int i = 0; i < cnt; ++i) {
      if (T < T_MIN_C) break;
      float tn = stn[wave * MAX_HITS + i];
      float tf = stf[wave * MAX_HITS + i];
      int   id = sid[wave * MAX_HITS + i];
      float dens = vox[(size_t)id * 8 + 6];
      float dt = (tf - tn) * (1.0f / S_SAMPLES_F);
      float op = 1.0f - expf(-dens * dt);
      float w = T * op;
      r  += w * colsig[id * 3 + 0];
      g  += w * colsig[id * 3 + 1];
      b  += w * colsig[id * 3 + 2];
      dep += w * (tn + tf) * 0.5f;
      T *= (1.0f - op);
    }
    out[ray * 3 + 0] = r;
    out[ray * 3 + 1] = g;
    out[ray * 3 + 2] = b;
    out[3 * N + ray] = dep;
    out[4 * N + ray] = 1.0f - T;
  }
}

// ---------------------------------------------------------------------------
extern "C" void kernel_launch(void* const* d_in, const int* in_sizes, int n_in,
                              void* d_out, int out_size, void* d_ws, size_t ws_size,
                              hipStream_t stream) {
  (void)n_in; (void)out_size; (void)ws_size;
  const float* positions = (const float*)d_in[0];
  const float* sizes     = (const float*)d_in[1];
  const float* densities = (const float*)d_in[2];
  const float* colors    = (const float*)d_in[3];
  const int*   morton    = (const int*)d_in[4];
  const float* ro        = (const float*)d_in[5];
  const float* rd        = (const float*)d_in[6];
  int V = in_sizes[1];
  int N = in_sizes[5] / 3;
  float* out = (float*)d_out;

  // Workspace layout (floats): [meand:8][keys:V][order:V][packed:8V][colsig:3V]
  float* wsf    = (float*)d_ws;
  float* meand  = wsf;
  float* keys   = wsf + 8;
  int*   order  = (int*)(wsf + 8 + V);
  float* packed = wsf + 8 + 2 * (size_t)V;
  float* colsig = packed + (size_t)V * 8;

  vr_meand<<<1, 256, 0, stream>>>(rd, meand, N);

  int waves   = (V + 15) / 16;
  int threads = waves * 32;
  int kblocks = (threads + 255) / 256;
  vr_keys<<<kblocks, 256, 0, stream>>>(positions, morton, meand, keys, V);

  vr_sort<<<1, 1024, 0, stream>>>(keys, order, V);

  vr_gather<<<(V + 255) / 256, 256, 0, stream>>>(order, positions, sizes,
                                                 densities, colors, packed,
                                                 colsig, V);

  int rblocks = (N + RAYS_PER_BLOCK - 1) / RAYS_PER_BLOCK;
  size_t smem = (size_t)V * 8 * sizeof(float)
              + (size_t)RAYS_PER_BLOCK * MAX_HITS * (2 * sizeof(float) + sizeof(int)) * 2;
  vr_render<<<rblocks, 256, smem, stream>>>(packed, colsig, ro, rd, out, V, N);
}
